// TransformerBlock_67980742361590
// MI455X (gfx1250) — compile-verified, matrix operand order checked by reading, not run on hardware
//
#include <hip/hip_runtime.h>
#include <math.h>

// ---------------------------------------------------------------------------
// Types for CDNA5 WMMA (wave32, 16x16x32 bf16 -> f32 accum)
// ---------------------------------------------------------------------------
typedef __attribute__((ext_vector_type(16))) __bf16 v16bf;
typedef __attribute__((ext_vector_type(8)))  __bf16 v8bf;
typedef __attribute__((ext_vector_type(8)))  float  v8f;

#define D_MODEL 1024
#define N_HEADS 16
#define D_FF    4096
#define D_K     64
#define SEQ     2048
#define BATCH   2
#define M_TOK   (BATCH * SEQ)   // 4096

// ---------------------------------------------------------------------------
// CDNA5 async global->LDS copy (ASYNCcnt tracked; no VGPR staging)
// ---------------------------------------------------------------------------
__device__ __forceinline__ void async_copy_b128(const __bf16* g, __bf16* lds) {
    asm volatile("global_load_async_to_lds_b128 %0, %1, off"
                 :
                 : "v"((__attribute__((address_space(3))) __bf16*)lds), "v"(g)
                 : "memory");
}

template <int N>
__device__ __forceinline__ void wait_asynccnt() {
    asm volatile("s_wait_asynccnt %0" : : "n"(N) : "memory");
}

// ---------------------------------------------------------------------------
// wave32 butterfly reductions + 8-wave block combine
// ---------------------------------------------------------------------------
__device__ __forceinline__ float wave_sum(float v) {
#pragma unroll
    for (int o = 16; o > 0; o >>= 1) v += __shfl_xor(v, o, 32);
    return v;
}
__device__ __forceinline__ float wave_max(float v) {
#pragma unroll
    for (int o = 16; o > 0; o >>= 1) v = fmaxf(v, __shfl_xor(v, o, 32));
    return v;
}

// ---------------------------------------------------------------------------
// f32 -> bf16 elementwise convert
// ---------------------------------------------------------------------------
__global__ __launch_bounds__(256) void conv_f32_bf16(const float* __restrict__ in,
                                                     __bf16* __restrict__ out, long n) {
    long i = (long)blockIdx.x * 256 + threadIdx.x;
    long stride = (long)gridDim.x * 256;
    for (; i < n; i += stride) out[i] = (__bf16)in[i];
}

// f32 [K][N] -> bf16 [N][K] (transposed weights so GEMMs can async-stage B)
__global__ __launch_bounds__(256)
void conv_transpose_f32_bf16(const float* __restrict__ in, __bf16* __restrict__ out,
                             int K, int N) {
    long i = (long)blockIdx.x * 256 + threadIdx.x;
    const long total = (long)K * N;
    const long stride = (long)gridDim.x * 256;
    for (; i < total; i += stride) {
        const long n = i / K, k = i % K;
        out[i] = (__bf16)in[k * (long)N + n];
    }
}

// ---------------------------------------------------------------------------
// Tiled WMMA GEMM: C[M,N] = A[M,K] * B (+bias, epilogue), double-buffered LDS,
// async global->LDS staging where the layout allows.
//   8 waves (256 threads); each wave computes (MW*16) x 64: MW A-frags x 4
//   B-frags -> 4*MW back-to-back WMMAs per 32-deep K step.
//   BM = MW*16*WROWS, BN = 64*WCOLS, WROWS*WCOLS == 8.
//   A_F32: A is f32 in memory, converted to bf16 while staging (manual path).
//   B_TRANS: B stored as [N][K] (contiguous k) -> async b128 staging.
//            otherwise B is [K][N] -> manual transpose staging.
//   EPI: 0 = store f32 * alpha (scores)
//        1 = (+bias if non-null) -> store bf16
//        2 = +bias -> store f32
//        3 = +bias -> exact GELU -> store bf16
//   Batch z: offset = (z>>4)*sHi + (z&15)*sLo per operand (H=16 inner dim).
// ---------------------------------------------------------------------------
template <int MW, int WROWS, int WCOLS, bool A_F32, bool B_TRANS, int EPI>
__global__ __launch_bounds__(256)
void gemm_wmma(const void* __restrict__ Av, const __bf16* __restrict__ Bg,
               const float* __restrict__ bias,
               float* __restrict__ Cf, __bf16* __restrict__ Cb,
               int M, int N, int Kdim, int lda, int ldb, int ldc,
               long sAhi, long sAlo, long sBhi, long sBlo, long sChi, long sClo,
               float alpha) {
    (void)M; (void)N;
    constexpr int BM   = MW * 16 * WROWS;
    constexpr int BN   = 64 * WCOLS;
    constexpr int KT   = 32;   // K per step (bf16 WMMA depth)
    constexpr int LDSK = 40;   // padded halfs per row: 80B, 16B-aligned, bank-skewed

    __shared__ __align__(16) __bf16 As[2 * BM * LDSK];
    __shared__ __align__(16) __bf16 Bs[2 * BN * LDSK];

    const int z = blockIdx.z;
    const long zA = (long)(z >> 4) * sAhi + (long)(z & 15) * sAlo;
    const long zB = (long)(z >> 4) * sBhi + (long)(z & 15) * sBlo;
    const long zC = (long)(z >> 4) * sChi + (long)(z & 15) * sClo;

    const __bf16* Ab = (const __bf16*)Av;  // valid when !A_F32
    const float*  Af = (const float*)Av;   // valid when A_F32
    const int tile_m = blockIdx.y * BM;
    const int tile_n = blockIdx.x * BN;

    const int tid  = threadIdx.x;
    const int lane = tid & 31;
    const int wave = tid >> 5;
    const int wr   = wave % WROWS;          // wave row   (MW*16 rows)
    const int wc   = wave / WROWS;          // wave col   (64 cols)

    // async b128 copies per thread per tile (BM*4, BN*4 divisible by 256)
    constexpr int APT = (A_F32 ? 0 : (BM * 4) / 256) + (B_TRANS ? (BN * 4) / 256 : 0);

    v8f acc[MW][4];
#pragma unroll
    for (int m = 0; m < MW; ++m)
#pragma unroll
        for (int c = 0; c < 4; ++c)
#pragma unroll
            for (int r = 0; r < 8; ++r) acc[m][c][r] = 0.0f;

    const int kb = (lane >> 4) * 8;         // per-lane K sub-chunk base

    auto stage = [&](int kt, int p) {
        __bf16* Asb = &As[p * BM * LDSK];
        __bf16* Bsb = &Bs[p * BN * LDSK];
        // ---- A tile: BM x 32 ----
        if (A_F32) {
            constexpr int CH = BM * 4;
            for (int i = tid; i < CH; i += 256) {
                const int row = i >> 2, ch = i & 3;
                const float* s = Af + zA + (long)(tile_m + row) * lda + kt + ch * 8;
                v8bf v;
#pragma unroll
                for (int j = 0; j < 8; ++j) v[j] = (__bf16)s[j];
                *(v8bf*)&Asb[row * LDSK + ch * 8] = v;
            }
        } else {
            constexpr int CH = BM * 4;
            for (int i = tid; i < CH; i += 256) {
                const int row = i >> 2, ch = i & 3;
                async_copy_b128(Ab + zA + (long)(tile_m + row) * lda + kt + ch * 8,
                                &Asb[row * LDSK + ch * 8]);
            }
        }
        // ---- B tile into [n][k] layout ----
        if (B_TRANS) {
            constexpr int CH = BN * 4;
            for (int i = tid; i < CH; i += 256) {
                const int nrow = i >> 2, ch = i & 3;
                async_copy_b128(Bg + zB + (long)(tile_n + nrow) * ldb + kt + ch * 8,
                                &Bsb[nrow * LDSK + ch * 8]);
            }
        } else {
            constexpr int CPR = BN / 8;
            constexpr int CH = 32 * CPR;
            for (int i = tid; i < CH; i += 256) {
                const int k = i / CPR, ch = i % CPR;
                v8bf v = *(const v8bf*)(Bg + zB + (long)(kt + k) * ldb + tile_n + ch * 8);
#pragma unroll
                for (int j = 0; j < 8; ++j) Bsb[(ch * 8 + j) * LDSK + k] = v[j];
            }
        }
    };

    // ---- double-buffered pipeline: stage t+1 while computing t ----
    stage(0, 0);
    int p = 0;
    for (int kt = 0; kt < Kdim; kt += KT, p ^= 1) {
        const bool nxt = (kt + KT < Kdim);
        if (nxt) stage(kt + KT, p ^ 1);
        if (APT > 0) {
            // ASYNC ops complete in order: <= APT outstanding => tile t done.
            if (nxt) wait_asynccnt<APT>();
            else     wait_asynccnt<0>();
        }
        __syncthreads();

        const __bf16* Asb = &As[p * BM * LDSK];
        const __bf16* Bsb = &Bs[p * BN * LDSK];
        // load all fragments first, then issue independent WMMAs back-to-back
        v16bf af[MW];
#pragma unroll
        for (int m = 0; m < MW; ++m) {
            const int mr = wr * (MW * 16) + m * 16 + (lane & 15);
            ((v8bf*)&af[m])[0] = *(const v8bf*)&Asb[mr * LDSK + kb];
            ((v8bf*)&af[m])[1] = *(const v8bf*)&Asb[mr * LDSK + kb + 16];
        }
        v16bf bf[4];
#pragma unroll
        for (int c = 0; c < 4; ++c) {
            const int n = wc * 64 + c * 16 + (lane & 15);
            ((v8bf*)&bf[c])[0] = *(const v8bf*)&Bsb[n * LDSK + kb];
            ((v8bf*)&bf[c])[1] = *(const v8bf*)&Bsb[n * LDSK + kb + 16];
        }
#pragma unroll
        for (int c = 0; c < 4; ++c)
#pragma unroll
            for (int m = 0; m < MW; ++m)
                acc[m][c] = __builtin_amdgcn_wmma_f32_16x16x32_bf16(
                    false, af[m], false, bf[c], (short)0, acc[m][c], false, false);
        __syncthreads();   // all reads of buffer p done before it is re-staged
    }

    // ---- epilogue ----
    const int lhalf = lane >> 4;
    const int lcol  = lane & 15;
#pragma unroll
    for (int m = 0; m < MW; ++m)
#pragma unroll
        for (int c = 0; c < 4; ++c) {
            const int col = tile_n + wc * 64 + c * 16 + lcol;
            const float bcol = (EPI != 0 && bias != nullptr) ? bias[col] : 0.0f;
#pragma unroll
            for (int r = 0; r < 8; ++r) {
                const int row = tile_m + wr * (MW * 16) + m * 16 + r + 8 * lhalf;
                const long off = zC + (long)row * ldc + col;
                float v = acc[m][c][r];
                if (EPI == 0) {
                    Cf[off] = v * alpha;
                } else {
                    v += bcol;
                    if (EPI == 1)      Cb[off] = (__bf16)v;
                    else if (EPI == 2) Cf[off] = v;
                    else { // exact GELU
                        const float g = 0.5f * v * (1.0f + erff(v * 0.70710678118654752f));
                        Cb[off] = (__bf16)g;
                    }
                }
            }
        }
}

// ---------------------------------------------------------------------------
// Row softmax, in place: one block per row of length 2048 (8 f32/thread)
// ---------------------------------------------------------------------------
__global__ __launch_bounds__(256) void softmax_rows(float* __restrict__ a, int n) {
    float* p = a + (long)blockIdx.x * n;
    const int tid = threadIdx.x;
    const int lane = tid & 31, wave = tid >> 5;
    float4 x0 = ((const float4*)p)[tid * 2];
    float4 x1 = ((const float4*)p)[tid * 2 + 1];
    __shared__ float red[8];

    float m = fmaxf(fmaxf(fmaxf(x0.x, x0.y), fmaxf(x0.z, x0.w)),
                    fmaxf(fmaxf(x1.x, x1.y), fmaxf(x1.z, x1.w)));
    m = wave_max(m);
    if (lane == 0) red[wave] = m;
    __syncthreads();
#pragma unroll
    for (int w = 0; w < 8; ++w) m = fmaxf(m, red[w]);
    __syncthreads();

    x0.x = expf(x0.x - m); x0.y = expf(x0.y - m); x0.z = expf(x0.z - m); x0.w = expf(x0.w - m);
    x1.x = expf(x1.x - m); x1.y = expf(x1.y - m); x1.z = expf(x1.z - m); x1.w = expf(x1.w - m);
    float s = x0.x + x0.y + x0.z + x0.w + x1.x + x1.y + x1.z + x1.w;
    s = wave_sum(s);
    if (lane == 0) red[wave] = s;
    __syncthreads();
    s = red[0] + red[1] + red[2] + red[3] + red[4] + red[5] + red[6] + red[7];

    const float inv = 1.0f / s;
    x0.x *= inv; x0.y *= inv; x0.z *= inv; x0.w *= inv;
    x1.x *= inv; x1.y *= inv; x1.z *= inv; x1.w *= inv;
    ((float4*)p)[tid * 2]     = x0;
    ((float4*)p)[tid * 2 + 1] = x1;
}

// ---------------------------------------------------------------------------
// LayerNorm(residual): out = LN(x + r) * g + b ; optional bf16 mirror copy
// One block per row of 1024 (4 f32/thread).
// ---------------------------------------------------------------------------
__global__ __launch_bounds__(256)
void layernorm_residual(const float* __restrict__ x, const float* __restrict__ r,
                        const float* __restrict__ g, const float* __restrict__ b,
                        float* __restrict__ outf, __bf16* __restrict__ outb, int n) {
    const long row = blockIdx.x;
    const int tid = threadIdx.x;
    const int lane = tid & 31, wave = tid >> 5;
    const float4 xv = ((const float4*)(x + row * n))[tid];
    const float4 rv = ((const float4*)(r + row * n))[tid];
    float4 s;
    s.x = xv.x + rv.x; s.y = xv.y + rv.y; s.z = xv.z + rv.z; s.w = xv.w + rv.w;

    __shared__ float red[8];
    float t = wave_sum(s.x + s.y + s.z + s.w);
    if (lane == 0) red[wave] = t;
    __syncthreads();
    t = red[0] + red[1] + red[2] + red[3] + red[4] + red[5] + red[6] + red[7];
    const float mu = t / (float)n;
    __syncthreads();

    float4 d;
    d.x = s.x - mu; d.y = s.y - mu; d.z = s.z - mu; d.w = s.w - mu;
    float q = wave_sum(d.x * d.x + d.y * d.y + d.z * d.z + d.w * d.w);
    if (lane == 0) red[wave] = q;
    __syncthreads();
    q = red[0] + red[1] + red[2] + red[3] + red[4] + red[5] + red[6] + red[7];
    const float inv = rsqrtf(q / (float)n + 1e-5f);

    const float4 gv = ((const float4*)g)[tid];
    const float4 bv = ((const float4*)b)[tid];
    float4 o;
    o.x = d.x * inv * gv.x + bv.x; o.y = d.y * inv * gv.y + bv.y;
    o.z = d.z * inv * gv.z + bv.z; o.w = d.w * inv * gv.w + bv.w;
    ((float4*)(outf + row * n))[tid] = o;
    if (outb) {
        __bf16* qd = outb + row * n + tid * 4;
        qd[0] = (__bf16)o.x; qd[1] = (__bf16)o.y; qd[2] = (__bf16)o.z; qd[3] = (__bf16)o.w;
    }
}

// ---------------------------------------------------------------------------
// Host orchestration
// ---------------------------------------------------------------------------
static inline dim3 g1d(long n) { return dim3((unsigned)((n + 255) / 256)); }

extern "C" void kernel_launch(void* const* d_in, const int* in_sizes, int n_in,
                              void* d_out, int out_size, void* d_ws, size_t ws_size,
                              hipStream_t stream) {
    (void)in_sizes; (void)n_in; (void)out_size; (void)ws_size;
    const float* x   = (const float*)d_in[0];
    const float* Wq  = (const float*)d_in[1];
    const float* bq  = (const float*)d_in[2];
    const float* Wk  = (const float*)d_in[3];
    const float* bk  = (const float*)d_in[4];
    const float* Wv  = (const float*)d_in[5];
    const float* bv  = (const float*)d_in[6];
    const float* Wo  = (const float*)d_in[7];
    const float* bo  = (const float*)d_in[8];
    const float* g1  = (const float*)d_in[9];
    const float* b1  = (const float*)d_in[10];
    const float* W1  = (const float*)d_in[11];
    const float* bf1 = (const float*)d_in[12];
    const float* W2  = (const float*)d_in[13];
    const float* bf2 = (const float*)d_in[14];
    const float* g2  = (const float*)d_in[15];
    const float* b2  = (const float*)d_in[16];

    float* out  = (float*)d_out;                         // [2,2048,1024]
    float* attn = out + (size_t)M_TOK * D_MODEL;         // [2,16,2048,2048]

    char* ws = (char*)d_ws;
    size_t off = 0;
    auto take = [&](size_t bytes) -> char* {
        char* p = ws + off;
        off = (off + bytes + 255) & ~(size_t)255;
        return p;
    };
    __bf16* xb  = (__bf16*)take((size_t)M_TOK * D_MODEL * 2);
    // transposed bf16 weights: [N][K] so GEMMs can async-stage B
    __bf16* Wqt = (__bf16*)take((size_t)D_MODEL * D_MODEL * 2);
    __bf16* Wkt = (__bf16*)take((size_t)D_MODEL * D_MODEL * 2);
    __bf16* Wvt = (__bf16*)take((size_t)D_MODEL * D_MODEL * 2);
    __bf16* Wot = (__bf16*)take((size_t)D_MODEL * D_MODEL * 2);
    __bf16* W1t = (__bf16*)take((size_t)D_MODEL * D_FF * 2);   // [4096][1024]
    __bf16* W2t = (__bf16*)take((size_t)D_FF * D_MODEL * 2);   // [1024][4096]
    char* qkv_region = ws + off;                         // aliased by gelub later
    __bf16* Qb   = (__bf16*)take((size_t)M_TOK * D_MODEL * 2);
    __bf16* Kb   = (__bf16*)take((size_t)M_TOK * D_MODEL * 2);
    __bf16* Vb   = (__bf16*)take((size_t)M_TOK * D_MODEL * 2);
    __bf16* ctxb = (__bf16*)take((size_t)M_TOK * D_MODEL * 2);
    __bf16* gelub = (__bf16*)qkv_region;                 // 32 MiB, reuses Q/K/V/ctx
    float* attnout = (float*)take((size_t)M_TOK * D_MODEL * 4);
    float* f2 = attnout;                                 // safe: attnout consumed by LN1
    float* h  = (float*)take((size_t)M_TOK * D_MODEL * 4);
    __bf16* hb = (__bf16*)take((size_t)M_TOK * D_MODEL * 2);

    // 1) bf16 conversions: activations plain, weights transposed
    conv_f32_bf16<<<g1d((long)M_TOK * D_MODEL), 256, 0, stream>>>(x, xb, (long)M_TOK * D_MODEL);
    conv_transpose_f32_bf16<<<g1d((long)D_MODEL * D_MODEL), 256, 0, stream>>>(Wq, Wqt, D_MODEL, D_MODEL);
    conv_transpose_f32_bf16<<<g1d((long)D_MODEL * D_MODEL), 256, 0, stream>>>(Wk, Wkt, D_MODEL, D_MODEL);
    conv_transpose_f32_bf16<<<g1d((long)D_MODEL * D_MODEL), 256, 0, stream>>>(Wv, Wvt, D_MODEL, D_MODEL);
    conv_transpose_f32_bf16<<<g1d((long)D_MODEL * D_MODEL), 256, 0, stream>>>(Wo, Wot, D_MODEL, D_MODEL);
    conv_transpose_f32_bf16<<<g1d((long)D_MODEL * D_FF), 256, 0, stream>>>(W1, W1t, D_MODEL, D_FF);
    conv_transpose_f32_bf16<<<g1d((long)D_FF * D_MODEL), 256, 0, stream>>>(W2, W2t, D_FF, D_MODEL);

    // 2) Q/K/V projections -> bf16 (128x128 tiles, fully async staged)
    gemm_wmma<2, 4, 2, false, true, 1><<<dim3(D_MODEL / 128, M_TOK / 128, 1), 256, 0, stream>>>(
        xb, Wqt, bq, nullptr, Qb, M_TOK, D_MODEL, D_MODEL, D_MODEL, D_MODEL, D_MODEL,
        0, 0, 0, 0, 0, 0, 1.0f);
    gemm_wmma<2, 4, 2, false, true, 1><<<dim3(D_MODEL / 128, M_TOK / 128, 1), 256, 0, stream>>>(
        xb, Wkt, bk, nullptr, Kb, M_TOK, D_MODEL, D_MODEL, D_MODEL, D_MODEL, D_MODEL,
        0, 0, 0, 0, 0, 0, 1.0f);
    gemm_wmma<2, 4, 2, false, true, 1><<<dim3(D_MODEL / 128, M_TOK / 128, 1), 256, 0, stream>>>(
        xb, Wvt, bv, nullptr, Vb, M_TOK, D_MODEL, D_MODEL, D_MODEL, D_MODEL, D_MODEL,
        0, 0, 0, 0, 0, 0, 1.0f);

    // 3) scores = (Q . K^T) / sqrt(64), z = b*16+h, f32 into d_out attn region
    const long sQKhi = (long)SEQ * D_MODEL;      // per batch
    const long sQKlo = D_K;                      // per head
    const long sAttnLo = (long)SEQ * SEQ;
    const long sAttnHi = 16 * sAttnLo;
    gemm_wmma<2, 4, 2, false, true, 0><<<dim3(SEQ / 128, SEQ / 128, BATCH * N_HEADS), 256, 0, stream>>>(
        Qb, Kb, nullptr, attn, nullptr, SEQ, SEQ, D_K, D_MODEL, D_MODEL, SEQ,
        sQKhi, sQKlo, sQKhi, sQKlo, sAttnHi, sAttnLo, 0.125f);

    // 4) softmax in place (this is the returned attention tensor)
    softmax_rows<<<dim3(BATCH * N_HEADS * SEQ), 256, 0, stream>>>(attn, SEQ);

    // 5) ctx = attn . V  (A f32 converted on stage-in; B = V [t][dk] manual)
    gemm_wmma<1, 8, 1, true, false, 1><<<dim3(1, SEQ / 128, BATCH * N_HEADS), 256, 0, stream>>>(
        attn, Vb, nullptr, nullptr, ctxb, SEQ, D_K, SEQ, SEQ, D_MODEL, D_MODEL,
        sAttnHi, sAttnLo, sQKhi, sQKlo, sQKhi, sQKlo, 1.0f);

    // 6) attn_out = ctx . Wo + bo -> f32
    gemm_wmma<2, 4, 2, false, true, 2><<<dim3(D_MODEL / 128, M_TOK / 128, 1), 256, 0, stream>>>(
        ctxb, Wot, bo, attnout, nullptr, M_TOK, D_MODEL, D_MODEL, D_MODEL, D_MODEL, D_MODEL,
        0, 0, 0, 0, 0, 0, 1.0f);

    // 7) h = LN(x + attn_out); keep f32 + bf16 mirror
    layernorm_residual<<<dim3(M_TOK), 256, 0, stream>>>(x, attnout, g1, b1, h, hb, D_MODEL);

    // 8) gelu(h . W1 + bf1) -> bf16
    gemm_wmma<2, 4, 2, false, true, 3><<<dim3(D_FF / 128, M_TOK / 128, 1), 256, 0, stream>>>(
        hb, W1t, bf1, nullptr, gelub, M_TOK, D_FF, D_MODEL, D_MODEL, D_MODEL, D_FF,
        0, 0, 0, 0, 0, 0, 1.0f);

    // 9) f2 = gelu . W2 + bf2 -> f32
    gemm_wmma<2, 4, 2, false, true, 2><<<dim3(D_MODEL / 128, M_TOK / 128, 1), 256, 0, stream>>>(
        gelub, W2t, bf2, f2, nullptr, M_TOK, D_MODEL, D_FF, D_FF, D_FF, D_MODEL,
        0, 0, 0, 0, 0, 0, 1.0f);

    // 10) out = LN(h + f2)
    layernorm_residual<<<dim3(M_TOK), 256, 0, stream>>>(h, f2, g2, b2, out, nullptr, D_MODEL);
}
